// HMModel_3350074490964
// MI455X (gfx1250) — matrix operand order repaired and checked by reading, not compiled, not run on hardware
//
#include <hip/hip_runtime.h>

// Problem constants from the reference.
constexpr int V  = 100000;  // vocab
constexpr int D  = 128;     // embedding dim
constexpr int B  = 32;      // batch
constexpr int H  = 50;      // history length
constexpr int HP = 64;      // history padded to multiple of 16 for WMMA M tiles

typedef __attribute__((ext_vector_type(16))) _Float16     v16h;
typedef __attribute__((ext_vector_type(8)))  _Float16     v8h;
typedef __attribute__((ext_vector_type(4)))  _Float16     v4h;
typedef __attribute__((ext_vector_type(8)))  float        v8f;
typedef __attribute__((ext_vector_type(4)))  float        v4f;
typedef __attribute__((ext_vector_type(8)))  unsigned int v8u;

// Cross-half (xor 16) shuffles. Always call these UNCONDITIONALLY before any
// divergent select so all 32 lanes participate.
__device__ __forceinline__ v16h shflx16_h(v16h x) {
  v8u u = __builtin_bit_cast(v8u, x);
#pragma unroll
  for (int j = 0; j < 8; ++j) u[j] = (unsigned)__shfl_xor((int)u[j], 16, 32);
  return __builtin_bit_cast(v16h, u);
}
__device__ __forceinline__ v8f shflx16_f(v8f x) {
#pragma unroll
  for (int j = 0; j < 8; ++j) x[j] = __shfl_xor(x[j], 16, 32);
  return x;
}
__device__ __forceinline__ float lrelu(float x) { return fmaxf(x, 0.01f * x); }

// ---------------------------------------------------------------------------
// Kernel 1: L2-normalize each embedding row (f32) -> f16 matrix wn_h [V, D].
// One wave per row; emb is read exactly once -> non-temporal load.
// ---------------------------------------------------------------------------
__global__ __launch_bounds__(256)
void hm_normalize_kernel(const float* __restrict__ emb, _Float16* __restrict__ wn) {
  const int wave = threadIdx.x >> 5;
  const int lane = threadIdx.x & 31;
  const int row  = blockIdx.x * 8 + wave;
  if (row >= V) return;

  const v4f x =
      __builtin_nontemporal_load(reinterpret_cast<const v4f*>(emb + (size_t)row * D) + lane);
  float ss = x[0] * x[0] + x[1] * x[1] + x[2] * x[2] + x[3] * x[3];
#pragma unroll
  for (int off = 16; off >= 1; off >>= 1) ss += __shfl_xor(ss, off, 32);
  const float inv = 1.0f / fmaxf(sqrtf(ss), 1e-12f);

  v4h o;
  o[0] = (_Float16)(x[0] * inv);
  o[1] = (_Float16)(x[1] * inv);
  o[2] = (_Float16)(x[2] * inv);
  o[3] = (_Float16)(x[3] * inv);
  reinterpret_cast<v4h*>(wn + (size_t)row * D)[lane] = o;
}

// ---------------------------------------------------------------------------
// Kernel 2: gather normalized history embeddings into xh_h [B, HP, D] (f16),
// rows h >= H zero-padded. One wave per (b, h_pad) row.
// ---------------------------------------------------------------------------
__global__ __launch_bounds__(256)
void hm_gather_hist_kernel(const int* __restrict__ hist,
                           const _Float16* __restrict__ wn,
                           _Float16* __restrict__ xh) {
  const int wave = threadIdx.x >> 5;
  const int lane = threadIdx.x & 31;
  const int row  = blockIdx.x * 8 + wave;   // row in [0, B*HP)
  if (row >= B * HP) return;
  const int b  = row >> 6;                  // HP == 64
  const int hp = row & 63;

  v4h o = {};
  if (hp < H) {
    const int idx = hist[b * H + hp];
    o = reinterpret_cast<const v4h*>(wn + (size_t)idx * D)[lane];
  }
  reinterpret_cast<v4h*>(xh + (size_t)row * D)[lane] = o;
}

// ---------------------------------------------------------------------------
// Kernel 3: each block owns a 32-wide vocab tile (two 16-wide WMMA N-tiles), its waves split batches.
// Per batch: WMMA scores [HP x 32] -> max/argmax over h -> logit/week/lik
// features -> MLP where layer 2 (8x32) also runs on WMMA.
//
// Fragment layouts used by this code (wave32, A and B both contiguous k):
//   A 16x32 f16: lane<16 -> row M=lane,   K elem e -> k = e
//                lane>=16 -> row M=lane-16, k = e + 16
//   B 32x16 f16: lane = n + 16*khalf -> column N=n, K = 16*khalf + e
//   C/D f32 16x16: lane -> N = lane%16, VGPR r -> M = r + 8*(lane/16)
// ---------------------------------------------------------------------------
__global__ __launch_bounds__(256)
void hm_score_mlp_kernel(const _Float16* __restrict__ wn,
                         const _Float16* __restrict__ xh,
                         const float* __restrict__ week,
                         const float* __restrict__ lik,
                         const float* __restrict__ w1, const float* __restrict__ b1,
                         const float* __restrict__ w2, const float* __restrict__ b2,
                         const float* __restrict__ w3, const float* __restrict__ b3,
                         float* __restrict__ out) {
  const int wave  = threadIdx.x >> 5;
  const int lane  = threadIdx.x & 31;
  const int n     = lane & 15;   // column within a 16-wide N tile / A row
  const int kh    = lane >> 4;   // K-half selector
  const int vbase = blockIdx.x * 32;

  // ---- Score-GEMM B fragments (both N tiles), reused across all batches.
  v16h bf0[4], bf1[4];
#pragma unroll
  for (int kc = 0; kc < 4; ++kc) {
    bf0[kc] = *reinterpret_cast<const v16h*>(wn + (size_t)(vbase + n) * D + kc * 32 + kh * 16);
    bf1[kc] = *reinterpret_cast<const v16h*>(wn + (size_t)(vbase + 16 + n) * D + kc * 32 + kh * 16);
  }

  const int   myv = vbase + lane;  // this lane's vocab column
  const float lk  = lik[myv];

  // ---- Layer-1: hoist the likelihood-dependent part (per-lane, loop-invariant).
  float base1[32];
#pragma unroll
  for (int o = 0; o < 32; ++o) base1[o] = fmaf(w1[o * 3 + 2], lk, b1[o]);

  // ---- Layer-2 W2 (8x32, zero-padded to 16x32) as a WMMA A-fragment (once).
  v16h a2;
#pragma unroll
  for (int e = 0; e < 16; ++e) {
    const int k = e + 16 * kh;
    a2[e] = (n < 8) ? (_Float16)w2[n * 32 + k] : (_Float16)0.0f;
  }
  // Bias b2 rides as the WMMA C input (rows 0-7 real, rows 8-15 pad).
  v8f c2init;
#pragma unroll
  for (int r = 0; r < 8; ++r) c2init[r] = (kh == 0) ? b2[r] : 0.0f;

  for (int bi = wave; bi < B; bi += 8) {
    const _Float16* xb = xh + (size_t)bi * HP * D;
    float rmax0 = -2.0f, rmax1 = -2.0f;  // unit-vector dots are in [-1, 1]
    int   ridx0 = 0,     ridx1 = 0;

    // ---- Score GEMM: [HP x 32] = xh[bi] (HPx128) x wn[vtile]^T.
#pragma unroll
    for (int mt = 0; mt < 4; ++mt) {
      v8f acc0 = {}, acc1 = {};
#pragma unroll
      for (int kc = 0; kc < 4; ++kc) {
        const _Float16* ap = xb + (size_t)(mt * 16 + n) * D + kc * 32 + kh * 16;
        const v8h lo = *reinterpret_cast<const v8h*>(ap);
        const v8h hi = *reinterpret_cast<const v8h*>(ap + 8);
        v16h a;
#pragma unroll
        for (int e = 0; e < 8; ++e) { a[e] = lo[e]; a[e + 8] = hi[e]; }
        acc0 = __builtin_amdgcn_wmma_f32_16x16x32_f16(false, a, false, bf0[kc],
                                                      (short)0, acc0, false, false);
        acc1 = __builtin_amdgcn_wmma_f32_16x16x32_f16(false, a, false, bf1[kc],
                                                      (short)0, acc1, false, false);
      }
      // Running max/argmax; h = mt*16 + kh*8 + r. For mt<3 always valid;
      // for mt==3 only kh==0, r<2 (h=48,49) can be valid.
#pragma unroll
      for (int r = 0; r < 8; ++r) {
        if (mt == 3 && r >= 2) continue;               // compile-time dead
        const int   h  = mt * 16 + kh * 8 + r;
        const bool  ok = (mt < 3) || (kh == 0);
        const float s0 = acc0[r];
        const float s1 = acc1[r];
        if (ok && s0 > rmax0) { rmax0 = s0; ridx0 = h; }
        if (ok && s1 > rmax1) { rmax1 = s1; ridx1 = h; }
      }
    }

    // ---- Cross-half argmax combine (prefer lower h on ties = jnp.argmax).
    {
      const float om = __shfl_xor(rmax0, 16, 32);
      const int   oi = __shfl_xor(ridx0, 16, 32);
      if (om > rmax0 || (om == rmax0 && oi < ridx0)) { rmax0 = om; ridx0 = oi; }
    }
    {
      const float om = __shfl_xor(rmax1, 16, 32);
      const int   oi = __shfl_xor(ridx1, 16, 32);
      if (om > rmax1 || (om == rmax1 && oi < ridx1)) { rmax1 = om; ridx1 = oi; }
    }
    // Lane ownership: lane L owns vocab column vbase + L.
    const float rmax = kh ? rmax1 : rmax0;
    const int   ridx = kh ? ridx1 : ridx0;

    // ---- Features.
    float s = fminf(fmaxf(rmax, 0.001f), 0.999f);
    s = -logf(1.0f / s - 1.0f);
    const float mw = week[bi * H + ridx];

    // ---- Layer 1 (3 -> 32) in VALU, emitted directly as packed f16.
    v16h plo, phi;
#pragma unroll
    for (int o = 0; o < 32; ++o) {
      float t = fmaf(w1[o * 3 + 0], s, fmaf(w1[o * 3 + 1], mw, base1[o]));
      t = lrelu(t);
      if (o < 16) plo[o] = (_Float16)t; else phi[o - 16] = (_Float16)t;
    }

    // ---- Layer 2 (32 -> 8) on WMMA. Build B fragments (K=32 feats x N=32
    // cols): tile0 cols = lanes 0-15, tile1 cols = lanes 16-31.
    const v16h sphi = shflx16_h(phi);   // unconditional cross-lane ops first
    const v16h splo = shflx16_h(plo);
    const v16h bt0 = kh ? sphi : plo;   // lane<16: own feats 0-15; lane>=16: feats 16-31 of lane-16
    const v16h bt1 = kh ? phi  : splo;  // lane<16: feats 0-15 of lane+16; lane>=16: own feats 16-31
    v8f h2a = __builtin_amdgcn_wmma_f32_16x16x32_f16(false, a2, false, bt0,
                                                     (short)0, c2init, false, false);
    v8f h2b = __builtin_amdgcn_wmma_f32_16x16x32_f16(false, a2, false, bt1,
                                                     (short)0, c2init, false, false);
    // h2a: lane c (<16) rows 0-7 = h2[0..8) of col vbase+c.
    // h2b: lane c (<16) rows 0-7 = h2[0..8) of col vbase+16+c -> shuffle up.
    const v8f h2bx = shflx16_f(h2b);
    const v8f h2   = kh ? h2bx : h2a;   // lane L: h2[0..8) of col vbase+L

    // ---- Layer 3 (8 -> 1).
    float t = b3[0];
#pragma unroll
    for (int r = 0; r < 8; ++r) t = fmaf(w3[r], lrelu(h2[r]), t);

    // Streaming output: non-temporal, fully coalesced (32 contiguous floats).
    __builtin_nontemporal_store(t, out + (size_t)bi * V + myv);
  }
}

// ---------------------------------------------------------------------------
// Launcher. Workspace layout: [wn_h: V*D f16 = 25,600,000 B][xh_h: B*HP*D f16
// = 524,288 B] -> ~26.2 MB total.
// ---------------------------------------------------------------------------
extern "C" void kernel_launch(void* const* d_in, const int* in_sizes, int n_in,
                              void* d_out, int out_size, void* d_ws, size_t ws_size,
                              hipStream_t stream) {
  const int*   article_hist = (const int*)  d_in[0];  // [B,H]
  const float* week_hist    = (const float*)d_in[1];  // [B,H]
  const float* emb_weight   = (const float*)d_in[2];  // [V,D]
  const float* likelihood   = (const float*)d_in[3];  // [V]
  const float* w1 = (const float*)d_in[4];
  const float* b1 = (const float*)d_in[5];
  const float* w2 = (const float*)d_in[6];
  const float* b2 = (const float*)d_in[7];
  const float* w3 = (const float*)d_in[8];
  const float* b3 = (const float*)d_in[9];
  float* out = (float*)d_out;                          // [B,V]

  _Float16* wn_h = (_Float16*)d_ws;
  _Float16* xh_h = (_Float16*)((char*)d_ws + (size_t)V * D * sizeof(_Float16));

  hm_normalize_kernel<<<V / 8, 256, 0, stream>>>(emb_weight, wn_h);
  hm_gather_hist_kernel<<<(B * HP) / 8, 256, 0, stream>>>(article_hist, wn_h, xh_h);
  hm_score_mlp_kernel<<<V / 32, 256, 0, stream>>>(
      wn_h, xh_h, week_hist, likelihood, w1, b1, w2, b2, w3, b3, out);
}
